// TemporalGNNRecommender_89481348645688
// MI455X (gfx1250) — matrix operand order repaired and checked by reading, not compiled, not run on hardware
//
#include <hip/hip_runtime.h>
#include <hip/hip_bf16.h>

#define NN 50000
#define NE 600000
#define EMBD 128
#define HIDD 64
#define NHEAD 4

typedef __attribute__((ext_vector_type(16))) __bf16 v16bf;
typedef __attribute__((ext_vector_type(8)))  float  v8f;

union Frag16 { v16bf v; unsigned u[8]; };

// ---------- transpose + convert weights: Wt[n*K + k] = bf16(W[k*N + n]) ----------
__global__ void wconvert_kernel(const float* __restrict__ W, __hip_bfloat16* __restrict__ Wt,
                                int K, int N) {
    int i = blockIdx.x * blockDim.x + threadIdx.x;
    if (i >= K * N) return;
    int n = i / K, k = i - n * K;
    Wt[i] = __float2bfloat16(W[(size_t)k * N + n]);
}

// ---------- embedding gather, f32 -> bf16 ----------
__global__ void gather_kernel(const int* __restrict__ x, const float* __restrict__ emb,
                              __hip_bfloat16* __restrict__ A) {
    long i = (long)blockIdx.x * blockDim.x + threadIdx.x;
    if (i >= (long)NN * EMBD) return;
    int node = (int)(i >> 7);
    int k = (int)(i & (EMBD - 1));
    A[i] = __float2bfloat16(emb[(size_t)x[node] * EMBD + k]);
}

// ---------- bf16 WMMA GEMM: H[M,N] = A[M,K] * B[K,N], B passed transposed [N,K] ----------
// block = 128 threads (4 waves), each wave owns a 16x16 tile; grid = (M/16, N/64)
template <int K>
__global__ void gemm_bf16_kernel(const __hip_bfloat16* __restrict__ A,
                                 const __hip_bfloat16* __restrict__ Bt,
                                 float* __restrict__ H, int N) {
    constexpr int Ku = K / 2;                 // packed dwords per row/col
    __shared__ unsigned ldsA[16 * Ku];        // 16 rows x K bf16
    const int tid = threadIdx.x;
    const int rowbase = blockIdx.x * 16;

    // stage A tile (row-major, contiguous) into LDS
    const unsigned* Ag = (const unsigned*)(A + (size_t)rowbase * K);
    for (int i = tid; i < 16 * Ku; i += 128) ldsA[i] = Ag[i];
    __syncthreads();

    const int wave = tid >> 5;
    const int lane = tid & 31;
    const int hi   = lane >> 4;               // half-wave select
    const int row  = lane & 15;
    const int colbase = blockIdx.y * 64 + wave * 16;
    const unsigned* Bg = (const unsigned*)Bt + (size_t)(colbase + row) * Ku;

    v8f c = {};
#pragma unroll
    for (int t = 0; t < K / 32; ++t) {
        Frag16 a, b;
        // A 16x32, contiguous k order used here (same as B): lanes 0-15 K=0..15, lanes 16-31 K=16..31
        const int ka = t * 32 + hi * 16;
#pragma unroll
        for (int j = 0; j < 8; ++j) a.u[j] = ldsA[row * Ku + (ka >> 1) + j];
        // B 32x16, contiguous k order used here (same as A): lanes 0-15 K=0..15 (N=lane), lanes 16-31 K=16..31 (N=lane-16)
        const int kb = t * 32 + hi * 16;
#pragma unroll
        for (int j = 0; j < 8; ++j) b.u[j] = Bg[(kb >> 1) + j];
        c = __builtin_amdgcn_wmma_f32_16x16x32_bf16(false, a.v, false, b.v,
                                                    (short)0, c, false, false);
    }
    // C layout: VGPR i -> M = i (lanes 0-15) / i+8 (lanes 16-31), N = lane&15
#pragma unroll
    for (int i = 0; i < 8; ++i) {
        int r = rowbase + i + hi * 8;
        H[(size_t)r * N + colbase + (lane & 15)] = c[i];
    }
}

// ---------- attention score dots: s[n,h] = <H[n, h*64 : h*64+64], a[h]> ----------
__global__ void scores_kernel(const float* __restrict__ H,
                              const float* __restrict__ a_s, const float* __restrict__ a_d,
                              float* __restrict__ ss, float* __restrict__ sd, int heads) {
    int i = blockIdx.x * blockDim.x + threadIdx.x;
    if (i >= NN * heads) return;
    int node = i / heads, h = i - node * heads;
    const float* hr = H + (size_t)node * heads * 64 + h * 64;
    const float* as = a_s + h * 64;
    const float* ad = a_d + h * 64;
    float s0 = 0.f, s1 = 0.f;
#pragma unroll 8
    for (int d = 0; d < 64; ++d) { float v = hr[d]; s0 += v * as[d]; s1 += v * ad[d]; }
    ss[i] = s0; sd[i] = s1;
}

// ---------- one wave per edge: denom[dst] += ex, acc[dst] += h[src]*ex ----------
__global__ void edge_aggregate_kernel(const float* __restrict__ H,
                                      const float* __restrict__ ss, const float* __restrict__ sd,
                                      const int* __restrict__ esrc, const int* __restrict__ edst,
                                      float* __restrict__ denom, float* __restrict__ acc,
                                      int heads, int C) {
    int wave = (blockIdx.x * blockDim.x + threadIdx.x) >> 5;
    int lane = threadIdx.x & 31;
    const int total = NE + NN;                // explicit edges + self loops
    if (wave >= total) return;
    int s, d;
    if (wave < NE) { s = esrc[wave]; d = edst[wave]; }
    else           { s = d = wave - NE; }

    float ex[NHEAD];
#pragma unroll
    for (int h = 0; h < NHEAD; ++h) {
        if (h < heads) {
            float e = ss[s * heads + h] + sd[d * heads + h];
            e = e > 0.f ? e : 0.2f * e;       // leaky_relu(0.2)
            ex[h] = __expf(e);                // softmax w/o max-shift (equivalent)
        } else ex[h] = 0.f;
    }
    if (lane < heads) atomicAdd(&denom[d * heads + lane], ex[lane]);

    const float* hs = H + (size_t)s * C;
    float* ao = acc + (size_t)d * C;
    for (int c = lane; c < C; c += 32)
        atomicAdd(&ao[c], hs[c] * ex[c >> 6]);
}

// ---------- normalize + bias + ELU; emit bf16 (next layer) or f32 (final) ----------
__global__ void finalize_kernel(const float* __restrict__ acc, const float* __restrict__ denom,
                                const float* __restrict__ bias,
                                float* __restrict__ out32, __hip_bfloat16* __restrict__ out16,
                                int heads, int C) {
    long i = (long)blockIdx.x * blockDim.x + threadIdx.x;
    if (i >= (long)NN * C) return;
    int node = (int)(i / C);
    int c = (int)(i - (long)node * C);
    float v = acc[i] / (denom[node * heads + (c >> 6)] + 1e-16f) + bias[c];
    v = v > 0.f ? v : (__expf(v) - 1.f);      // ELU
    if (out32) out32[i] = v;
    else       out16[i] = __float2bfloat16(v);
}

extern "C" void kernel_launch(void* const* d_in, const int* in_sizes, int n_in,
                              void* d_out, int out_size, void* d_ws, size_t ws_size,
                              hipStream_t stream) {
    const int*   x    = (const int*)  d_in[0];
    const int*   ei   = (const int*)  d_in[1];   // [2, NE]: row0 = src, row1 = dst
    const float* emb  = (const float*)d_in[3];
    const float* W1   = (const float*)d_in[4];
    const float* as1  = (const float*)d_in[5];
    const float* ad1  = (const float*)d_in[6];
    const float* b1   = (const float*)d_in[7];
    const float* W2   = (const float*)d_in[8];
    const float* as2  = (const float*)d_in[9];
    const float* ad2  = (const float*)d_in[10];
    const float* b2   = (const float*)d_in[11];
    const int* esrc = ei;
    const int* edst = ei + NE;

    // ---- carve workspace (256B aligned), with lifetime-based aliasing ----
    char* ws = (char*)d_ws;
    size_t off = 0;
    auto take = [&](size_t bytes) -> void* {
        void* p = ws + off;
        off += (bytes + 255) & ~(size_t)255;
        return p;
    };
    __hip_bfloat16* W1t = (__hip_bfloat16*)take((size_t)256 * 128 * 2);
    __hip_bfloat16* W2t = (__hip_bfloat16*)take((size_t)64 * 256 * 2);
    __hip_bfloat16* A1  = (__hip_bfloat16*)take((size_t)NN * EMBD * 2);     // 12.8 MB
    float* H1   = (float*)take((size_t)NN * 256 * 4);                        // 51.2 MB
    float* ss1  = (float*)take((size_t)NN * NHEAD * 4);
    float* sd1  = (float*)take((size_t)NN * NHEAD * 4);
    float* dn1  = (float*)take((size_t)NN * NHEAD * 4);
    float* acc1 = (float*)take((size_t)NN * 256 * 4);                        // 51.2 MB
    float* ss2  = (float*)take((size_t)NN * 4);
    float* sd2  = (float*)take((size_t)NN * 4);
    float* dn2  = (float*)take((size_t)NN * 4);
    // aliases (non-overlapping lifetimes):
    __hip_bfloat16* A2 = (__hip_bfloat16*)H1;   // written after H1 last read
    float* H2   = acc1;                          // written after acc1 last read
    float* acc2 = (float*)A1;                    // A1 dead after GEMM1; same size

    // ---- layer 0: weight prep + gather ----
    wconvert_kernel<<<(128 * 256 + 255) / 256, 256, 0, stream>>>(W1, W1t, 128, 256);
    wconvert_kernel<<<(256 * 64 + 255) / 256, 256, 0, stream>>>(W2, W2t, 256, 64);
    gather_kernel<<<((long)NN * EMBD + 255) / 256, 256, 0, stream>>>(x, emb, A1);

    // ---- layer 1: GEMM (bf16 WMMA), scores, edge scatter, finalize ----
    gemm_bf16_kernel<128><<<dim3(NN / 16, 256 / 64), 128, 0, stream>>>(A1, W1t, H1, 256);
    scores_kernel<<<(NN * NHEAD + 255) / 256, 256, 0, stream>>>(H1, as1, ad1, ss1, sd1, NHEAD);
    hipMemsetAsync(dn1, 0, (size_t)NN * NHEAD * 4, stream);
    hipMemsetAsync(acc1, 0, (size_t)NN * 256 * 4, stream);
    {
        int waves = NE + NN;
        edge_aggregate_kernel<<<(waves * 32 + 255) / 256, 256, 0, stream>>>(
            H1, ss1, sd1, esrc, edst, dn1, acc1, NHEAD, 256);
    }
    finalize_kernel<<<((long)NN * 256 + 255) / 256, 256, 0, stream>>>(
        acc1, dn1, b1, nullptr, A2, NHEAD, 256);

    // ---- layer 2 ----
    gemm_bf16_kernel<256><<<dim3(NN / 16, 1), 128, 0, stream>>>(A2, W2t, H2, 64);
    scores_kernel<<<(NN + 255) / 256, 256, 0, stream>>>(H2, as2, ad2, ss2, sd2, 1);
    hipMemsetAsync(dn2, 0, (size_t)NN * 4, stream);
    hipMemsetAsync(acc2, 0, (size_t)NN * HIDD * 4, stream);
    {
        int waves = NE + NN;
        edge_aggregate_kernel<<<(waves * 32 + 255) / 256, 256, 0, stream>>>(
            H2, ss2, sd2, esrc, edst, dn2, acc2, 1, HIDD);
    }
    finalize_kernel<<<((long)NN * HIDD + 255) / 256, 256, 0, stream>>>(
        acc2, dn2, b2, (float*)d_out, nullptr, 1, HIDD);
    (void)in_sizes; (void)n_in; (void)out_size; (void)ws_size;
}
